// MultiPolicyRLAttention_6485400617592
// MI455X (gfx1250) — matrix operand order repaired and checked by reading, not compiled, not run on hardware
//
#include <hip/hip_runtime.h>
#include <cstddef>
#include <cstdint>

// ---------------------------------------------------------------------------
// Types for CDNA5 WMMA (wave32): A/B = 16 bf16 per lane (16x32 / 32x16 tile),
// C/D = 8 f32 per lane (16x16 tile).
// ---------------------------------------------------------------------------
typedef __attribute__((ext_vector_type(16))) __bf16 v16bf;
typedef __attribute__((ext_vector_type(8)))  float  v8f;
typedef __attribute__((ext_vector_type(4)))  unsigned int u32x4;
typedef __attribute__((ext_vector_type(8)))  int  i32x8;
typedef __attribute__((ext_vector_type(4)))  int  i32x4;

union AF {            // 32 bytes: one WMMA A or B fragment
    uint4  u[2];
    v16bf  v;
};

// CDNA5 Tensor Data Mover available? (amdgpu-toolchain clang-23: 6-arg builtin)
#if defined(__gfx1250__) && __has_builtin(__builtin_amdgcn_tensor_load_to_lds) && \
    __has_builtin(__builtin_amdgcn_s_wait_tensorcnt)
#define USE_TDM 1
#else
#define USE_TDM 0
#endif

__device__ __forceinline__ unsigned short f32_to_bf16(float f) {
    unsigned int u = __float_as_uint(f);
    unsigned int r = u + 0x7FFFu + ((u >> 16) & 1u);   // round-to-nearest-even
    return (unsigned short)(r >> 16);
}

__device__ __forceinline__ v8f wmma_bf16(const AF& a, const AF& b, v8f c) {
    return __builtin_amdgcn_wmma_f32_16x16x32_bf16(
        false, a.v, false, b.v, (short)0, c, false, false);
}

// 32 contiguous bytes global -> 32 contiguous bytes LDS, ASYNCcnt-tracked.
// (GLOBAL_LOAD_ASYNC_TO_LDS_B128: 16 bytes global -> LDS per lane, same offset)
__device__ __forceinline__ void async_copy32(unsigned lds_addr, const void* gptr) {
    asm volatile("global_load_async_to_lds_b128 %0, %1, off\n\t"
                 "global_load_async_to_lds_b128 %0, %1, off offset:16"
                 :: "v"(lds_addr), "v"(gptr) : "memory");
}
__device__ __forceinline__ void wait_async0() {
    asm volatile("s_wait_asynccnt 0" ::: "memory");
}
__device__ __forceinline__ void wait_ds0() {
    asm volatile("s_wait_dscnt 0" ::: "memory");
}

// ---------------------------------------------------------------------------
// fp32 -> bf16 conversion
// ---------------------------------------------------------------------------
__global__ void cvt_bf16_kernel(const float* __restrict__ src,
                                unsigned short* __restrict__ dst, int n) {
    int i = blockIdx.x * blockDim.x + threadIdx.x;
    if (i < n) dst[i] = f32_to_bf16(src[i]);
}

// ---------------------------------------------------------------------------
// mean pool over L:  xm[b,d] = mean_l x[b,l,d]
// ---------------------------------------------------------------------------
__global__ void meanpool_kernel(const float* __restrict__ x,
                                float* __restrict__ xm,
                                int Bn, int Ln, int Dn) {
    int i = blockIdx.x * blockDim.x + threadIdx.x;
    if (i >= Bn * Dn) return;
    int b = i / Dn, d = i % Dn;
    const float* p = x + (size_t)b * Ln * Dn + d;
    float s = 0.f;
    for (int l = 0; l < Ln; ++l) s += p[(size_t)l * Dn];
    xm[i] = s / (float)Ln;
}

// ---------------------------------------------------------------------------
// policy hidden layers: s_hid = gelu(xm @ sw1^T + sb1)   [B, DH]
//                       p_hid = gelu(xm @ pw1   + pb1)   [B, P, DH]
// ---------------------------------------------------------------------------
__global__ void policy_hidden_kernel(const float* __restrict__ xm,
                                     const float* __restrict__ sw1,
                                     const float* __restrict__ sb1,
                                     const float* __restrict__ pw1,
                                     const float* __restrict__ pb1,
                                     float* __restrict__ s_hid,
                                     float* __restrict__ p_hid,
                                     int Bn, int Dn, int DHn, int Pn) {
    int i = blockIdx.x * blockDim.x + threadIdx.x;
    int total = Bn * (1 + Pn) * DHn;
    if (i >= total) return;
    int b   = i / ((1 + Pn) * DHn);
    int r   = i % ((1 + Pn) * DHn);
    int net = r / DHn;
    int c   = r % DHn;
    const float* xv = xm + (size_t)b * Dn;
    const float* wrow;
    float acc;
    if (net == 0) { wrow = sw1 + (size_t)c * Dn; acc = sb1[c]; }
    else {
        int p = net - 1;
        wrow = pw1 + ((size_t)p * DHn + c) * Dn;
        acc  = pb1[p * DHn + c];
    }
    for (int d = 0; d < Dn; ++d) acc += xv[d] * wrow[d];
    float gel = 0.5f * acc * (1.0f + erff(acc * 0.70710678118654752f));
    if (net == 0) s_hid[(size_t)b * DHn + c] = gel;
    else          p_hid[((size_t)b * Pn + (net - 1)) * DHn + c] = gel;
}

// ---------------------------------------------------------------------------
// policy output layers + softmaxes + mixture -> head_gates[B, H]
// ---------------------------------------------------------------------------
__global__ void policy_gates_kernel(const float* __restrict__ s_hid,
                                    const float* __restrict__ p_hid,
                                    const float* __restrict__ sw2,
                                    const float* __restrict__ sb2,
                                    const float* __restrict__ pw2,
                                    const float* __restrict__ pb2,
                                    float* __restrict__ head_gates,
                                    int DHn, int Pn, int Hn) {
    __shared__ float sel[8];
    __shared__ float gl[64];
    int b = blockIdx.x, t = threadIdx.x;
    if (t < Pn) {
        const float* hv = s_hid + (size_t)b * DHn;
        const float* w  = sw2 + (size_t)t * DHn;
        float a = sb2[t];
        for (int d = 0; d < DHn; ++d) a += hv[d] * w[d];
        sel[t] = a;
    }
    int nt = t - Pn;
    if (nt >= 0 && nt < Pn * Hn) {
        int p = nt / Hn, hh = nt % Hn;
        const float* hv = p_hid + ((size_t)b * Pn + p) * DHn;
        const float* w  = pw2 + ((size_t)p * Hn + hh) * DHn;
        float a = pb2[p * Hn + hh];
        for (int d = 0; d < DHn; ++d) a += hv[d] * w[d];
        gl[p * Hn + hh] = a;
    }
    __syncthreads();
    if (t == 0) {
        float wv[3];
        float mx = sel[0];
        for (int p = 1; p < Pn; ++p) mx = fmaxf(mx, sel[p]);
        float sw = 0.f;
        for (int p = 0; p < Pn; ++p) { wv[p] = __expf(sel[p] - mx); sw += wv[p]; }
        for (int p = 0; p < Pn; ++p) wv[p] /= sw;
        float hg[16];
        for (int i = 0; i < Hn; ++i) hg[i] = 0.f;
        for (int p = 0; p < Pn; ++p) {
            float m2 = gl[p * Hn];
            for (int i = 1; i < Hn; ++i) m2 = fmaxf(m2, gl[p * Hn + i]);
            float s2 = 0.f, e[16];
            for (int i = 0; i < Hn; ++i) { e[i] = __expf(gl[p * Hn + i] - m2); s2 += e[i]; }
            for (int i = 0; i < Hn; ++i) hg[i] += wv[p] * e[i] / s2;
        }
        for (int i = 0; i < Hn; ++i) head_gates[(size_t)b * Hn + i] = hg[i];
    }
}

// ---------------------------------------------------------------------------
// Generic WMMA GEMM:  C[m,n] = sum_k A[m,k] * W[n,k] + bias[n]
// A: [M,K] bf16 (K contig).  W: [N,K] bf16 (K contig).
// Block: 256 thr (8 waves), tile 128x128, K step 32.
// A tile staged via GLOBAL_LOAD_ASYNC_TO_LDS_B128 (ASYNCcnt path);
// W tile staged via TENSOR_LOAD_TO_LDS issued by wave 0 (TENSORcnt path),
// falling back to async copies when the TDM builtin is unavailable.
// mode 0: bf16 Q [B,H,L,64] (x0.125) | 1: bf16 K [B,H,L,64]
// mode 2: bf16 V^T [B,H,64,L]        | 3: f32 out [M,N]
// ---------------------------------------------------------------------------
__global__ void __launch_bounds__(256)
gemm_bf16_kernel(const unsigned short* __restrict__ A,
                 const unsigned short* __restrict__ W,
                 const float* __restrict__ bias,
                 void* __restrict__ out,
                 int M, int N, int K, int mode,
                 int Bn, int Ln, int Hn) {
    __shared__ __align__(16) unsigned short As[128 * 32];
    __shared__ __align__(16) unsigned short Bs[128 * 32];

    const int tid  = threadIdx.x;
    const int w    = tid >> 5;
    const int lane = tid & 31;
    const int r    = lane & 15;
    const int hk   = lane >> 4;
    const int wr   = w >> 1;          // 0..3 : 32-row strip
    const int wc   = w & 1;           // 0..1 : 64-col strip
    const int bx   = blockIdx.x;      // N / 128
    const int by   = blockIdx.y;      // M / 128

    const int ldrow = tid >> 1;       // 0..127
    const int ldseg = tid & 1;        // 0..1  (32 bytes each)

    // per-thread staging addresses (32B contiguous on both sides)
    const unsigned short* gA = A + (size_t)(by * 128 + ldrow) * K + ldseg * 16;
    const unsigned lA = (unsigned)(uintptr_t)(As + ldrow * 32 + ldseg * 16);
#if !USE_TDM
    const unsigned short* gB = W + (size_t)(bx * 128 + ldrow) * K + ldseg * 16;
    const unsigned lB = (unsigned)(uintptr_t)(Bs + ldrow * 32 + ldseg * 16);
#endif

    v8f acc[2][4];
#pragma unroll
    for (int mt = 0; mt < 2; ++mt)
#pragma unroll
        for (int nt = 0; nt < 4; ++nt)
            acc[mt][nt] = (v8f){0.f, 0.f, 0.f, 0.f, 0.f, 0.f, 0.f, 0.f};

    for (int k0 = 0; k0 < K; k0 += 32) {
        // ---- stage A tile: per-lane direct global->LDS async copies ------
        async_copy32(lA, gA + k0);

        // ---- stage W tile ------------------------------------------------
#if USE_TDM
        if (w == 0) {
            // 2-D TDM descriptor: 128 rows x 64B, row stride K*2 bytes,
            // data_size = 8B units.
            unsigned long long gaddr =
                (unsigned long long)(uintptr_t)(W + (size_t)(bx * 128) * K) +
                (unsigned long long)k0 * 2ull;
            u32x4 g0 = {
                1u,                                        // count=1, user D#
                (unsigned)(uintptr_t)Bs,                   // lds_addr
                (unsigned)(gaddr & 0xFFFFFFFFu),           // global_addr[31:0]
                (unsigned)((gaddr >> 32) & 0x1FFFFFFu) | (2u << 30) // [56:32]|type=2
            };
            const int rowU = (K * 2) / 8;                  // row stride/len in 8B units
            i32x8 g1 = {
                (int)(3u << 16),                           // data_size=8B
                (int)((unsigned)(rowU & 0xFFFF) << 16),    // tensor_dim0 lo16
                (int)(128u << 16),                         // dim0 hi=0 | tensor_dim1 lo16=128
                (int)(8u << 16),                           // dim1 hi=0 | tile_dim0=8 units
                (int)128,                                  // tile_dim1=128, tile_dim2=0
                (int)rowU,                                 // tensor_dim0_stride lo32
                0, 0
            };
            i32x4 gz4 = {0, 0, 0, 0};
            i32x8 gz8 = {0, 0, 0, 0, 0, 0, 0, 0};
            // amdgpu-toolchain (clang-23) 6-arg form:
            // (g0, g1, g2, g3, <extra>, cpol)
            __builtin_amdgcn_tensor_load_to_lds(g0, g1, gz4, gz4, gz8, 0);
            __builtin_amdgcn_s_wait_tensorcnt(0);
        }
#else
        async_copy32(lB, gB + k0);
#endif
        if (k0 + 32 < K)    // near-cache hint for the next A strip
            __builtin_prefetch(gA + k0 + 32, 0, 3);

        wait_async0();
        __syncthreads();

        // ---- fragments (all loads hoisted ahead of the WMMA burst) -------
        AF af[2], bfr[4];
#pragma unroll
        for (int mt = 0; mt < 2; ++mt) {
            const uint4* pa = (const uint4*)(As + (wr * 32 + mt * 16 + r) * 32 + hk * 16);
            af[mt].u[0] = pa[0];          // K = 16*hk .. 16*hk+7
            af[mt].u[1] = pa[1];          // K = 16*hk+8 .. 16*hk+15
        }
#pragma unroll
        for (int nt = 0; nt < 4; ++nt) {
            const uint4* pb = (const uint4*)(Bs + (wc * 64 + nt * 16 + r) * 32 + hk * 16);
            bfr[nt].u[0] = pb[0];         // K = 16*hk .. 16*hk+15
            bfr[nt].u[1] = pb[1];
        }
#pragma unroll
        for (int nt = 0; nt < 4; ++nt)
#pragma unroll
            for (int mt = 0; mt < 2; ++mt)
                acc[mt][nt] = wmma_bf16(af[mt], bfr[nt], acc[mt][nt]);
        __syncthreads();
    }

    // epilogue: C element (M = v + 8*hk, N = r) within each 16x16 tile
#pragma unroll
    for (int mt = 0; mt < 2; ++mt) {
#pragma unroll
        for (int nt = 0; nt < 4; ++nt) {
#pragma unroll
            for (int v = 0; v < 8; ++v) {
                int gm = by * 128 + wr * 32 + mt * 16 + v + 8 * hk;
                int gn = bx * 128 + wc * 64 + nt * 16 + r;
                float val = acc[mt][nt][v] + bias[gn];
                if (mode == 3) {
                    ((float*)out)[(size_t)gm * N + gn] = val;
                } else {
                    int bb = gm / Ln, ll = gm % Ln;
                    int hh = gn >> 6, hd = gn & 63;
                    unsigned short* o = (unsigned short*)out;
                    if (mode == 0) {
                        o[(((size_t)(bb * Hn + hh)) * Ln + ll) * 64 + hd] =
                            f32_to_bf16(val * 0.125f);
                    } else if (mode == 1) {
                        o[(((size_t)(bb * Hn + hh)) * Ln + ll) * 64 + hd] =
                            f32_to_bf16(val);
                    } else {
                        o[(((size_t)(bb * Hn + hh)) * 64 + hd) * Ln + ll] =
                            f32_to_bf16(val);
                    }
                }
            }
        }
    }
}

// ---------------------------------------------------------------------------
// Flash attention with per-head gate g and causal mask:
//   softmax over k of [ masked(QK^T/8) * g ] @ V
// Q pre-scaled by 0.125. Block = 8 waves; wave owns 16 query rows.
// ---------------------------------------------------------------------------
__global__ void __launch_bounds__(256)
flash_attn_kernel(const unsigned short* __restrict__ Qb,
                  const unsigned short* __restrict__ Kb,
                  const unsigned short* __restrict__ Vtb,
                  const float* __restrict__ head_gates,
                  unsigned short* __restrict__ aout,
                  int Bn, int Hn, int Ln) {
    __shared__ __align__(16) unsigned short Plds[8 * 16 * 32];  // per-wave 1KB

    const int tid  = threadIdx.x;
    const int w    = tid >> 5;
    const int lane = tid & 31;
    const int r    = lane & 15;
    const int hk   = lane >> 4;

    const int qtiles = Ln >> 7;                       // 128-row q blocks
    const int qb = blockIdx.x % qtiles;
    const int hh = (blockIdx.x / qtiles) % Hn;
    const int b  = blockIdx.x / (qtiles * Hn);
    const int q0 = qb * 128 + w * 16;

    const float g = head_gates[(size_t)b * Hn + hh];
    const size_t bhL = ((size_t)(b * Hn + hh)) * Ln;

    // Q A-fragments (HD split into two K=32 chunks)
    AF qa[2];
    {
        const unsigned short* qrow = Qb + (bhL + (q0 + r)) * 64;
#pragma unroll
        for (int j = 0; j < 2; ++j) {
            const uint4* p0 = (const uint4*)(qrow + j * 32 + hk * 16);
            const uint4* p1 = (const uint4*)(qrow + j * 32 + hk * 16 + 8);
            qa[j].u[0] = *p0;
            qa[j].u[1] = *p1;
        }
    }

    v8f O0 = {0.f,0.f,0.f,0.f,0.f,0.f,0.f,0.f};
    v8f O1 = O0, O2 = O0, O3 = O0;
    float mrow[8], srow[8];
#pragma unroll
    for (int v = 0; v < 8; ++v) { mrow[v] = -3.0e38f; srow[v] = 0.f; }

    const int kbmax = (q0 + 15) >> 5;
    for (int kb = 0; kb <= kbmax; ++kb) {
        // ---- scores for 16q x 32k strip : two 16x16 C frags --------------
        v8f s0 = {0.f,0.f,0.f,0.f,0.f,0.f,0.f,0.f};
        v8f s1 = s0;
        {
            const unsigned short* krow = Kb + (bhL + (kb * 32 + r)) * 64;
            AF kf0, kf1;
            const uint4* p0 = (const uint4*)(krow + hk * 16);
            const uint4* p1 = (const uint4*)(krow + 32 + hk * 16);
            kf0.u[0] = p0[0]; kf0.u[1] = p0[1];
            kf1.u[0] = p1[0]; kf1.u[1] = p1[1];
            s0 = wmma_bf16(qa[0], kf0, s0);
            s0 = wmma_bf16(qa[1], kf1, s0);
        }
        {
            const unsigned short* krow = Kb + (bhL + (kb * 32 + 16 + r)) * 64;
            AF kf0, kf1;
            const uint4* p0 = (const uint4*)(krow + hk * 16);
            const uint4* p1 = (const uint4*)(krow + 32 + hk * 16);
            kf0.u[0] = p0[0]; kf0.u[1] = p0[1];
            kf1.u[0] = p1[0]; kf1.u[1] = p1[1];
            s1 = wmma_bf16(qa[0], kf0, s1);
            s1 = wmma_bf16(qa[1], kf1, s1);
        }

        // ---- gate + causal mask + online softmax -------------------------
        float p0v[8], p1v[8], alpha[8];
#pragma unroll
        for (int v = 0; v < 8; ++v) {
            int qr  = q0 + v + 8 * hk;
            int kc  = kb * 32 + r;
            float v0 = (kc      > qr) ? -1.0e9f * g : g * s0[v];
            float v1 = (kc + 16 > qr) ? -1.0e9f * g : g * s1[v];
            float mx = fmaxf(v0, v1);
#pragma unroll
            for (int off = 1; off < 16; off <<= 1)
                mx = fmaxf(mx, __shfl_xor(mx, off, 32));
            float mn = fmaxf(mrow[v], mx);
            float al = __expf(mrow[v] - mn);
            mrow[v] = mn;
            alpha[v] = al;
            float e0 = __expf(v0 - mn), e1 = __expf(v1 - mn);
            p0v[v] = e0; p1v[v] = e1;
            float rs = e0 + e1;
#pragma unroll
            for (int off = 1; off < 16; off <<= 1)
                rs += __shfl_xor(rs, off, 32);
            srow[v] = srow[v] * al + rs;
        }
#pragma unroll
        for (int v = 0; v < 8; ++v) {
            O0[v] *= alpha[v]; O1[v] *= alpha[v];
            O2[v] *= alpha[v]; O3[v] *= alpha[v];
        }

        // ---- repack P (C layout -> A layout) via wave-private LDS --------
        unsigned short* pw = Plds + w * 512;
#pragma unroll
        for (int v = 0; v < 8; ++v) {
            int row = v + 8 * hk;
            pw[row * 32 + r]      = f32_to_bf16(p0v[v]);
            pw[row * 32 + r + 16] = f32_to_bf16(p1v[v]);
        }
        wait_ds0();                                   // wave-local RAW fence
        AF pf;
        {
            const uint4* pa = (const uint4*)(pw + r * 32 + hk * 16);
            pf.u[0] = pa[0];
            pf.u[1] = pa[1];
        }

        // ---- O += P @ V (V^T rows are key-contiguous) --------------------
#pragma unroll
        for (int t = 0; t < 4; ++t) {
            int hd = t * 16 + r;
            const unsigned short* vrow =
                Vtb + (((size_t)(b * Hn + hh)) * 64 + hd) * Ln + kb * 32 + hk * 16;
            AF vf;
            const uint4* p = (const uint4*)vrow;
            vf.u[0] = p[0]; vf.u[1] = p[1];
            if      (t == 0) O0 = wmma_bf16(pf, vf, O0);
            else if (t == 1) O1 = wmma_bf16(pf, vf, O1);
            else if (t == 2) O2 = wmma_bf16(pf, vf, O2);
            else             O3 = wmma_bf16(pf, vf, O3);
        }
    }

    // ---- normalize + store bf16 attention output [B*L, D] ----------------
    const int Dn = Hn * 64;
#pragma unroll
    for (int v = 0; v < 8; ++v) {
        float inv = 1.0f / srow[v];
        int q = q0 + v + 8 * hk;
        size_t base = ((size_t)(b * Ln + q)) * Dn + hh * 64;
        aout[base +  0 + r] = f32_to_bf16(O0[v] * inv);
        aout[base + 16 + r] = f32_to_bf16(O1[v] * inv);
        aout[base + 32 + r] = f32_to_bf16(O2[v] * inv);
        aout[base + 48 + r] = f32_to_bf16(O3[v] * inv);
    }
}

// ---------------------------------------------------------------------------
// launcher
// ---------------------------------------------------------------------------
extern "C" void kernel_launch(void* const* d_in, const int* in_sizes, int n_in,
                              void* d_out, int out_size, void* d_ws, size_t ws_size,
                              hipStream_t stream) {
    (void)in_sizes; (void)n_in; (void)out_size; (void)ws_size;
    const int B = 2, L = 2048, D = 1024, H = 16, P = 3, DH = 512;

    const float* x   = (const float*)d_in[0];
    const float* wq  = (const float*)d_in[1];
    const float* bq  = (const float*)d_in[2];
    const float* wk  = (const float*)d_in[3];
    const float* bk  = (const float*)d_in[4];
    const float* wv  = (const float*)d_in[5];
    const float* bv  = (const float*)d_in[6];
    const float* wo  = (const float*)d_in[7];
    const float* bo  = (const float*)d_in[8];
    const float* pw1 = (const float*)d_in[9];
    const float* pb1 = (const float*)d_in[10];
    const float* pw2 = (const float*)d_in[11];
    const float* pb2 = (const float*)d_in[12];
    const float* sw1 = (const float*)d_in[13];
    const float* sb1 = (const float*)d_in[14];
    const float* sw2 = (const float*)d_in[15];
    const float* sb2 = (const float*)d_in[16];

    char* wsp = (char*)d_ws;
    size_t off = 0;
    auto alloc = [&](size_t bytes) -> void* {
        void* p = wsp + off;
        off += (bytes + 255) & ~(size_t)255;
        return p;
    };
    unsigned short* xbf  = (unsigned short*)alloc((size_t)B * L * D * 2);
    unsigned short* wqb  = (unsigned short*)alloc((size_t)D * D * 2);
    unsigned short* wkb  = (unsigned short*)alloc((size_t)D * D * 2);
    unsigned short* wvb  = (unsigned short*)alloc((size_t)D * D * 2);
    unsigned short* wob  = (unsigned short*)alloc((size_t)D * D * 2);
    unsigned short* Qbf  = (unsigned short*)alloc((size_t)B * L * D * 2);
    unsigned short* Kbf  = (unsigned short*)alloc((size_t)B * L * D * 2);
    unsigned short* Vtbf = (unsigned short*)alloc((size_t)B * L * D * 2);
    unsigned short* aobf = (unsigned short*)alloc((size_t)B * L * D * 2);
    float* xm    = (float*)alloc((size_t)B * D * 4);
    float* s_hid = (float*)alloc((size_t)B * DH * 4);
    float* p_hid = (float*)alloc((size_t)B * P * DH * 4);
    float* hg    = (float*)alloc((size_t)B * H * 4);

    // 1) conversions
    int n;
    n = B * L * D; cvt_bf16_kernel<<<(n + 255) / 256, 256, 0, stream>>>(x,  xbf, n);
    n = D * D;     cvt_bf16_kernel<<<(n + 255) / 256, 256, 0, stream>>>(wq, wqb, n);
    cvt_bf16_kernel<<<(n + 255) / 256, 256, 0, stream>>>(wk, wkb, n);
    cvt_bf16_kernel<<<(n + 255) / 256, 256, 0, stream>>>(wv, wvb, n);
    cvt_bf16_kernel<<<(n + 255) / 256, 256, 0, stream>>>(wo, wob, n);

    // 2) policy gating (tiny)
    meanpool_kernel<<<(B * D + 255) / 256, 256, 0, stream>>>(x, xm, B, L, D);
    n = B * (1 + P) * DH;
    policy_hidden_kernel<<<(n + 255) / 256, 256, 0, stream>>>(
        xm, sw1, sb1, pw1, pb1, s_hid, p_hid, B, D, DH, P);
    policy_gates_kernel<<<B, 64, 0, stream>>>(
        s_hid, p_hid, sw2, sb2, pw2, pb2, hg, DH, P, H);

    // 3) QKV projections (WMMA GEMM, async-LDS + TDM staging)
    dim3 gg(D / 128, (B * L) / 128);
    gemm_bf16_kernel<<<gg, 256, 0, stream>>>(xbf, wqb, bq, Qbf,  B * L, D, D, 0, B, L, H);
    gemm_bf16_kernel<<<gg, 256, 0, stream>>>(xbf, wkb, bk, Kbf,  B * L, D, D, 1, B, L, H);
    gemm_bf16_kernel<<<gg, 256, 0, stream>>>(xbf, wvb, bv, Vtbf, B * L, D, D, 2, B, L, H);

    // 4) gated causal flash attention (WMMA)
    flash_attn_kernel<<<B * H * (L / 128), 256, 0, stream>>>(
        Qbf, Kbf, Vtbf, hg, aobf, B, H, L);

    // 5) output projection -> fp32 d_out
    gemm_bf16_kernel<<<gg, 256, 0, stream>>>(aobf, wob, bo, d_out, B * L, D, D, 3, B, L, H);
}
